// EquivariantMPNNModel_13649406067048
// MI455X (gfx1250) — compile-verified, matrix operand order checked by reading, not run on hardware
//
#include <hip/hip_runtime.h>
#include <cstdint>

// ---------------------------------------------------------------------------
// EGNN-style MPNN on MI455X (gfx1250, wave32, WMMA).
// GEMMs: tall-skinny (rows x K) x (K x 64) via v_wmma_f32_16x16x32_bf16,
// fp32 accumulate. All E x 64 / N x 64 intermediates are stored as bf16
// (halves HBM traffic vs fp32 — the dominant cost at 23.3 TB/s).
// C tiles are restaged through LDS so stores are coalesced 128B bf16 rows.
// BN: stats accumulated in fp32 from WMMA accumulators (ds_add_f32 + one
// global_atomic_add_f32 per channel per block); affine+ReLU fused into the
// next stage's A-tile build. Biases feeding training-mode BN are no-ops.
// gemm_tile_body MUST be force-inlined: an outlined copy gets generic
// pointers -> flat_load_b128 + cndmask address selects (observed in r2).
// ---------------------------------------------------------------------------

#define NN  50000
#define EE  800000
#define GG  64
#define DD  64
#define DINN 11
#define LLY 2

typedef __attribute__((ext_vector_type(16))) __bf16 v16bf;
typedef __attribute__((ext_vector_type(8)))  __bf16 v8bf;
typedef __attribute__((ext_vector_type(8)))  float  v8f;

// ----------------------------- fragment loads ------------------------------
// A-matrix 16x32 bf16 fragment per ISA layout:
//   lane 0-15: M=lane,  halfs 0-7 -> K=kk..kk+7,    halfs 8-15 -> K=kk+16..kk+23
//   lane16-31: M=lane-16, halfs 0-7 -> K=kk+8..+15, halfs 8-15 -> K=kk+24..+31
// => two contiguous 16B LDS chunks (ds_load_b128 pair).
__device__ __forceinline__ v16bf load_a_frag(const __bf16* ldsA, int Kpad,
                                             int mtile, int kk, int lane) {
  const __bf16* r = ldsA + (mtile + (lane & 15)) * Kpad + kk + ((lane >> 4) << 3);
  v8bf lo = *(const v8bf*)r;
  v8bf hi = *(const v8bf*)(r + 16);
  v16bf f;
#pragma unroll
  for (int j = 0; j < 8; ++j) { f[j] = lo[j]; f[8 + j] = hi[j]; }
  return f;
}

// B fragments pre-swizzled in global memory: one 32B load per lane (L2-hot).
__device__ __forceinline__ v16bf load_b_frag(const v16bf* Bswz, int kk4, int nt,
                                             int lane) {
  return Bswz[(size_t)(kk4 + nt) * 32 + lane];
}

// ------------------------- 128x64 WMMA tile body ---------------------------
// blockDim == 128 (4 waves). Wave w computes rows [w*32, w*32+32) x all 64
// cols. Output: bf16 Cout rows (coalesced via LDS restage), fp32 BN stats.
// ldsAB: A tile on entry; reused as bf16 C tile (128*64) after the K loop.
__device__ __forceinline__ void gemm_tile_body(
    __bf16* ldsAB, int Kpad, const unsigned short* Bswz_u,
    unsigned short* __restrict__ Cout_u, long long tileRow, long long rowsTotal,
    float* __restrict__ gsum, float* __restrict__ gsqs,
    float* ldsSum, float* ldsSqs) {
  const v16bf* Bswz = (const v16bf*)Bswz_u;
  int lane = threadIdx.x & 31;
  int wave = threadIdx.x >> 5;
  int m0 = wave * 32;

  v8f acc[2][4] = {};
  for (int kk = 0; kk < Kpad; kk += 32) {
    v16bf a0 = load_a_frag(ldsAB, Kpad, m0, kk, lane);
    v16bf a1 = load_a_frag(ldsAB, Kpad, m0 + 16, kk, lane);
    int kk4 = (kk >> 5) << 2;
#pragma unroll
    for (int nt = 0; nt < 4; ++nt) {
      v16bf b = load_b_frag(Bswz, kk4, nt, lane);
      acc[0][nt] = __builtin_amdgcn_wmma_f32_16x16x32_bf16(
          false, a0, false, b, (short)0, acc[0][nt], false, false);
      acc[1][nt] = __builtin_amdgcn_wmma_f32_16x16x32_bf16(
          false, a1, false, b, (short)0, acc[1][nt], false, false);
    }
  }
  __syncthreads();  // all waves done reading A tile; reuse LDS for C tile

  // C layout: lane -> N = lane&15, VGPR i -> M = i + 8*(lane>>4)
  __bf16* ldsC = ldsAB;
  int colBase = lane & 15;
  int rowOff = (lane >> 4) << 3;
#pragma unroll
  for (int h = 0; h < 2; ++h) {
#pragma unroll
    for (int nt = 0; nt < 4; ++nt) {
      int col = nt * 16 + colBase;
      float ps = 0.f, pq = 0.f;
#pragma unroll
      for (int i = 0; i < 8; ++i) {
        int r = m0 + h * 16 + rowOff + i;
        float v = acc[h][nt][i];
        ldsC[r * 64 + col] = (__bf16)v;
        if (tileRow + r < rowsTotal) { ps += v; pq += v * v; }
      }
      unsafeAtomicAdd(&ldsSum[col], ps);   // ds_add_f32
      unsafeAtomicAdd(&ldsSqs[col], pq);
    }
  }
  __syncthreads();

  // Coalesced store: each thread writes its contiguous 128B bf16 row.
  long long row = tileRow + threadIdx.x;
  if (row < rowsTotal) {
    v8bf* dp = (v8bf*)(Cout_u + row * 64);
    const v8bf* sp = (const v8bf*)(ldsC + threadIdx.x * 64);
#pragma unroll
    for (int q = 0; q < 8; ++q) dp[q] = sp[q];
  }
  if (threadIdx.x < 64) {
    unsafeAtomicAdd(&gsum[threadIdx.x], ldsSum[threadIdx.x]);
    unsafeAtomicAdd(&gsqs[threadIdx.x], ldsSqs[threadIdx.x]);
  }
}

// ------------------------------- utilities ---------------------------------
__global__ void k_zero(float* p, long long n) {
  long long i = (long long)blockIdx.x * blockDim.x + threadIdx.x;
  if (i < n) p[i] = 0.f;
}

__global__ void k_copy(float* d, const float* s, long long n) {
  long long i = (long long)blockIdx.x * blockDim.x + threadIdx.x;
  if (i < n) d[i] = s[i];
}

__global__ void k_deg(const int* dst, float* deg) {
  int e = blockIdx.x * blockDim.x + threadIdx.x;
  if (e < EE) unsafeAtomicAdd(&deg[dst[e]], 1.f);
}

// Swizzle (K x 64) fp32 weight into per-lane bf16 B-fragment order, zero-pad K.
__global__ void k_prep_w(const float* W, int K, int Kpad, unsigned short* out_u) {
  __bf16* out = (__bf16*)out_u;
  int idx = blockIdx.x * blockDim.x + threadIdx.x;
  if (idx >= Kpad * 64) return;
  int j = idx & 15;
  int lane = (idx >> 4) & 31;
  int t = idx >> 9;
  int nt = t & 3;
  int kkblk = t >> 2;
  int k = kkblk * 32 + j + 8 * ((j >> 3) + (lane >> 4));
  int n = nt * 16 + (lane & 15);
  float v = (k < K) ? W[k * 64 + n] : 0.f;
  out[idx] = (__bf16)v;
}

// h = x @ lin_in_w + lin_in_b  (N x 11 -> N x 64)
__global__ void k_lin_in(const float* x, const float* w, const float* b, float* h) {
  __shared__ float ws[DINN * 64];
  __shared__ float bs[64];
  int tid = threadIdx.x;
  for (int i = tid; i < DINN * 64; i += blockDim.x) ws[i] = w[i];
  if (tid < 64) bs[tid] = b[tid];
  __syncthreads();
  int node = blockIdx.x * blockDim.x + tid;
  if (node >= NN) return;
  float xv[DINN];
#pragma unroll
  for (int k = 0; k < DINN; ++k) xv[k] = x[node * DINN + k];
  float* hr = h + (size_t)node * 64;
  for (int c = 0; c < 64; ++c) {
    float a = bs[c];
#pragma unroll
    for (int k = 0; k < DINN; ++k) a += xv[k] * ws[k * 64 + c];
    hr[c] = a;
  }
}

// scale/shift finalize: y = x*scale + shift  ==  gamma*(x-mean)*rsqrt(var)+beta
__global__ void k_bnfin(const float* gsum, const float* gsqs, const float* gam,
                        const float* bet, float cnt, float* scale, float* shift) {
  int c = threadIdx.x;
  if (c >= 64) return;
  float mean = gsum[c] / cnt;
  float var = gsqs[c] / cnt - mean * mean;
  float sc = gam[c] * __frsqrt_rn(var + 1e-5f);
  scale[c] = sc;
  shift[c] = bet[c] - mean * sc;
}

// ---------------------- stage 1: edge concat -> msg_w1 ---------------------
__global__ void k_gemm_edge1(const float* __restrict__ h, const float* __restrict__ p,
                             const float* __restrict__ eattr, const int* src,
                             const int* dst, const unsigned short* Bswz,
                             unsigned short* Cout, float* gsum, float* gsqs) {
  __shared__ __align__(16) __bf16 ldsA[128 * 160];
  __shared__ float ldsSum[64], ldsSqs[64];
  int tid = threadIdx.x;
  if (tid < 64) { ldsSum[tid] = 0.f; ldsSqs[tid] = 0.f; }
  long long e = (long long)blockIdx.x * 128 + tid;  // E % 128 == 0
  int si = src[e], di = dst[e];
  const float* hi = h + (size_t)di * 64;
  const float* hj = h + (size_t)si * 64;
  __bf16* ar = ldsA + tid * 160;
#pragma unroll
  for (int c = 0; c < 64; ++c) { ar[c] = (__bf16)hi[c]; ar[64 + c] = (__bf16)hj[c]; }
  float r2 = 0.f;
#pragma unroll
  for (int k = 0; k < 3; ++k) {
    float pd = p[di * 3 + k] - p[si * 3 + k];
    r2 += pd * pd;
  }
  ar[128] = (__bf16)r2;
#pragma unroll
  for (int k = 0; k < 4; ++k) ar[129 + k] = (__bf16)eattr[e * 4 + k];
#pragma unroll
  for (int k = 133; k < 160; ++k) ar[k] = (__bf16)0.f;
  __syncthreads();
  gemm_tile_body(ldsA, 160, Bswz, Cout, (long long)blockIdx.x * 128, EE,
                 gsum, gsqs, ldsSum, ldsSqs);
}

// ------------- generic stage: A = relu(T*scale+shift), K=64 GEMM -----------
__global__ void k_gemm_act(const unsigned short* __restrict__ Tin_u,
                           const float* scale, const float* shift,
                           const unsigned short* Bswz, unsigned short* Cout,
                           float* gsum, float* gsqs, long long rowsTotal) {
  __shared__ __align__(16) __bf16 ldsA[128 * 64];
  __shared__ float ldsSum[64], ldsSqs[64];
  int tid = threadIdx.x;
  if (tid < 64) { ldsSum[tid] = 0.f; ldsSqs[tid] = 0.f; }
  long long row = (long long)blockIdx.x * 128 + tid;
  __bf16* ar = ldsA + tid * 64;
  if (row < rowsTotal) {
    const __bf16* t = (const __bf16*)Tin_u + row * 64;
#pragma unroll
    for (int c = 0; c < 64; ++c) {
      float v = (float)t[c] * scale[c] + shift[c];
      v = v > 0.f ? v : 0.f;
      ar[c] = (__bf16)v;
    }
  } else {
#pragma unroll
    for (int c = 0; c < 64; ++c) ar[c] = (__bf16)0.f;
  }
  __syncthreads();
  gemm_tile_body(ldsA, 64, Bswz, Cout, (long long)blockIdx.x * 128, rowsTotal,
                 gsum, gsqs, ldsSum, ldsSqs);
}

// same, plus scatter-add of m = relu(bn(t2)) into m_aggr[dst] (segment_sum)
__global__ void k_gemm_act_scatter(const unsigned short* __restrict__ Tin_u,
                                   const float* scale, const float* shift,
                                   const unsigned short* Bswz, unsigned short* Cout,
                                   float* gsum, float* gsqs, const int* dst,
                                   float* __restrict__ maggr) {
  __shared__ __align__(16) __bf16 ldsA[128 * 64];
  __shared__ float ldsSum[64], ldsSqs[64];
  int tid = threadIdx.x;
  if (tid < 64) { ldsSum[tid] = 0.f; ldsSqs[tid] = 0.f; }
  long long row = (long long)blockIdx.x * 128 + tid;  // always < EE
  const __bf16* t = (const __bf16*)Tin_u + row * 64;
  float* mg = maggr + (size_t)dst[row] * 64;
  __bf16* ar = ldsA + tid * 64;
#pragma unroll
  for (int c = 0; c < 64; ++c) {
    float v = (float)t[c] * scale[c] + shift[c];
    v = v > 0.f ? v : 0.f;
    ar[c] = (__bf16)v;
    unsafeAtomicAdd(&mg[c], v);  // global_atomic_add_f32 (maggr L2-resident)
  }
  __syncthreads();
  gemm_tile_body(ldsA, 64, Bswz, Cout, (long long)blockIdx.x * 128, EE,
                 gsum, gsqs, ldsSum, ldsSqs);
}

// ----------------- node update stage 1: A=[h | m_aggr], K=128 --------------
__global__ void k_gemm_upd1(const float* __restrict__ h, const float* __restrict__ maggr,
                            const unsigned short* Bswz, unsigned short* Cout,
                            float* gsum, float* gsqs) {
  __shared__ __align__(16) __bf16 ldsA[128 * 128];
  __shared__ float ldsSum[64], ldsSqs[64];
  int tid = threadIdx.x;
  if (tid < 64) { ldsSum[tid] = 0.f; ldsSqs[tid] = 0.f; }
  long long row = (long long)blockIdx.x * 128 + tid;
  __bf16* ar = ldsA + tid * 128;
  if (row < NN) {
    const float* hr = h + row * 64;
    const float* mr = maggr + row * 64;
#pragma unroll
    for (int c = 0; c < 64; ++c) { ar[c] = (__bf16)hr[c]; ar[64 + c] = (__bf16)mr[c]; }
  } else {
#pragma unroll
    for (int c = 0; c < 128; ++c) ar[c] = (__bf16)0.f;
  }
  __syncthreads();
  gemm_tile_body(ldsA, 128, Bswz, Cout, (long long)blockIdx.x * 128, NN,
                 gsum, gsqs, ldsSum, ldsSqs);
}

// --------- pos gate: s = relu(bn3(t3)) @ pos_w2 + b2;  pacc += pd*s --------
__global__ void k_pos_gate(const unsigned short* __restrict__ t3_u,
                           const float* scale, const float* shift,
                           const float* w2, const float* b2,
                           const float* __restrict__ p, const int* src,
                           const int* dst, float* __restrict__ pacc) {
  __shared__ float w2s[64];
  int tid = threadIdx.x;
  if (tid < 64) w2s[tid] = w2[tid];
  __syncthreads();
  long long e = (long long)blockIdx.x * blockDim.x + tid;
  if (e >= EE) return;
  const __bf16* t = (const __bf16*)t3_u + e * 64;
  float s = b2[0];
#pragma unroll
  for (int c = 0; c < 64; ++c) {
    float v = (float)t[c] * scale[c] + shift[c];
    v = v > 0.f ? v : 0.f;
    s += v * w2s[c];
  }
  int si = src[e], di = dst[e];
#pragma unroll
  for (int k = 0; k < 3; ++k) {
    float pd = p[di * 3 + k] - p[si * 3 + k];
    unsafeAtomicAdd(&pacc[di * 3 + k], pd * s);
  }
}

// --------- final node update: h += relu(bn5(t5)); p += pacc/deg ------------
__global__ void k_node_final(const unsigned short* __restrict__ t5_u,
                             const float* scale, const float* shift,
                             float* __restrict__ h, float* __restrict__ p,
                             const float* pacc, const float* deg) {
  int i = blockIdx.x * blockDim.x + threadIdx.x;
  if (i >= NN) return;
  const __bf16* t = (const __bf16*)t5_u + (size_t)i * 64;
  float* hr = h + (size_t)i * 64;
#pragma unroll
  for (int c = 0; c < 64; ++c) {
    float v = (float)t[c] * scale[c] + shift[c];
    hr[c] += v > 0.f ? v : 0.f;
  }
  float d = deg[i];
  d = d < 1.f ? 1.f : d;
#pragma unroll
  for (int k = 0; k < 3; ++k) p[i * 3 + k] += pacc[i * 3 + k] / d;
}

// --------------------------- pooling + prediction --------------------------
__global__ void k_pool(const float* __restrict__ h, const int* batch,
                       float* __restrict__ hg, float* __restrict__ cnt) {
  int i = blockIdx.x * blockDim.x + threadIdx.x;
  if (i >= NN) return;
  int g = batch[i];
  const float* hr = h + (size_t)i * 64;
  float* gr = hg + (size_t)g * 64;
#pragma unroll
  for (int c = 0; c < 64; ++c) unsafeAtomicAdd(&gr[c], hr[c]);
  unsafeAtomicAdd(&cnt[g], 1.f);
}

__global__ void k_out(const float* hg, const float* cnt, const float* w,
                      const float* b, float* out) {
  int g = threadIdx.x;
  if (g >= GG) return;
  float c = cnt[g];
  c = c < 1.f ? 1.f : c;
  float a = b[0];
  const float* r = hg + (size_t)g * 64;
#pragma unroll
  for (int k = 0; k < 64; ++k) a += (r[k] / c) * w[k];
  out[g] = a;
}

// ------------------------------- launcher ----------------------------------
extern "C" void kernel_launch(void* const* d_in, const int* in_sizes, int n_in,
                              void* d_out, int out_size, void* d_ws, size_t ws_size,
                              hipStream_t stream) {
  const float* x        = (const float*)d_in[0];
  const float* pos      = (const float*)d_in[1];
  const float* eattr    = (const float*)d_in[2];
  const int*   ei       = (const int*)d_in[3];
  const int*   batch    = (const int*)d_in[4];
  const float* lin_in_w = (const float*)d_in[5];
  const float* lin_in_b = (const float*)d_in[6];
  const float* pred_w   = (const float*)d_in[7];
  const float* pred_b   = (const float*)d_in[8];
  const float* msg_w1   = (const float*)d_in[9];
  const float* msg_g1   = (const float*)d_in[11];
  const float* msg_be1  = (const float*)d_in[12];
  const float* msg_w2   = (const float*)d_in[13];
  const float* msg_g2   = (const float*)d_in[15];
  const float* msg_be2  = (const float*)d_in[16];
  const float* pos_w1   = (const float*)d_in[17];
  const float* pos_g    = (const float*)d_in[19];
  const float* pos_be   = (const float*)d_in[20];
  const float* pos_w2   = (const float*)d_in[21];
  const float* pos_b2   = (const float*)d_in[22];
  const float* upd_w1   = (const float*)d_in[23];
  const float* upd_g1   = (const float*)d_in[25];
  const float* upd_be1  = (const float*)d_in[26];
  const float* upd_w2   = (const float*)d_in[27];
  const float* upd_g2   = (const float*)d_in[29];
  const float* upd_be2  = (const float*)d_in[30];
  const int* src = ei;
  const int* dst = ei + EE;

  // ---- workspace carve-out (256B-aligned) ----
  char* w = (char*)d_ws;
  auto carve = [&](size_t bytes) -> char* {
    char* r = w;
    w += (bytes + 255) & ~(size_t)255;
    return r;
  };
  float* h     = (float*)carve((size_t)NN * 64 * 4);
  float* p     = (float*)carve((size_t)NN * 3 * 4);
  float* deg   = (float*)carve((size_t)NN * 4);
  float* maggr = (float*)carve((size_t)NN * 64 * 4);
  float* pacc  = (float*)carve((size_t)NN * 3 * 4);
  float* hg    = (float*)carve((size_t)GG * 64 * 4 + (size_t)GG * 4);  // hg|cnt
  float* cnt   = hg + GG * 64;
  float* ssum  = (float*)carve(128 * 4);  // ssum|ssqs
  float* ssqs  = ssum + 64;
  float* scale = (float*)carve(128 * 4);  // scale|shift
  float* shift = scale + 64;
  unsigned short* bufA = (unsigned short*)carve((size_t)EE * 64 * 2);  // bf16
  unsigned short* bufB = (unsigned short*)carve((size_t)EE * 64 * 2);  // bf16
  unsigned short* wm1[LLY], *wm2[LLY], *wp1[LLY], *wu1[LLY], *wu2[LLY];
  for (int l = 0; l < LLY; ++l) {
    wm1[l] = (unsigned short*)carve(160 * 64 * 2);
    wm2[l] = (unsigned short*)carve(64 * 64 * 2);
    wp1[l] = (unsigned short*)carve(64 * 64 * 2);
    wu1[l] = (unsigned short*)carve(128 * 64 * 2);
    wu2[l] = (unsigned short*)carve(64 * 64 * 2);
  }

  const int EB = EE / 128;          // 6250 edge tiles (exact)
  const int NB = (NN + 127) / 128;  // 391 node tiles
  auto gz = [](long long n) { return (unsigned)((n + 255) / 256); };

  // ---- init ----
  k_zero<<<gz(NN), 256, 0, stream>>>(deg, NN);
  k_deg<<<gz(EE), 256, 0, stream>>>(dst, deg);
  k_lin_in<<<NB, 128, 0, stream>>>(x, lin_in_w, lin_in_b, h);
  k_copy<<<gz((long long)NN * 3), 256, 0, stream>>>(p, pos, (long long)NN * 3);
  for (int l = 0; l < LLY; ++l) {
    k_prep_w<<<gz(160 * 64), 256, 0, stream>>>(msg_w1 + (size_t)l * 133 * 64, 133, 160, wm1[l]);
    k_prep_w<<<gz(64 * 64), 256, 0, stream>>>(msg_w2 + (size_t)l * 64 * 64, 64, 64, wm2[l]);
    k_prep_w<<<gz(64 * 64), 256, 0, stream>>>(pos_w1 + (size_t)l * 64 * 64, 64, 64, wp1[l]);
    k_prep_w<<<gz(128 * 64), 256, 0, stream>>>(upd_w1 + (size_t)l * 128 * 64, 128, 128, wu1[l]);
    k_prep_w<<<gz(64 * 64), 256, 0, stream>>>(upd_w2 + (size_t)l * 64 * 64, 64, 64, wu2[l]);
  }

  // ---- layers ----
  for (int l = 0; l < LLY; ++l) {
    k_zero<<<gz((long long)NN * 64), 256, 0, stream>>>(maggr, (long long)NN * 64);
    k_zero<<<gz((long long)NN * 3), 256, 0, stream>>>(pacc, (long long)NN * 3);

    // S1: t1 = concat(h_i,h_j,r2,eattr) @ msg_w1  (+stats)
    k_zero<<<1, 128, 0, stream>>>(ssum, 128);
    k_gemm_edge1<<<EB, 128, 0, stream>>>(h, p, eattr, src, dst, wm1[l], bufA, ssum, ssqs);
    k_bnfin<<<1, 64, 0, stream>>>(ssum, ssqs, msg_g1 + l * 64, msg_be1 + l * 64,
                                  (float)EE, scale, shift);
    // S3: t2 = relu(bn1(t1)) @ msg_w2  (+stats)
    k_zero<<<1, 128, 0, stream>>>(ssum, 128);
    k_gemm_act<<<EB, 128, 0, stream>>>(bufA, scale, shift, wm2[l], bufB, ssum, ssqs, EE);
    k_bnfin<<<1, 64, 0, stream>>>(ssum, ssqs, msg_g2 + l * 64, msg_be2 + l * 64,
                                  (float)EE, scale, shift);
    // S5: m = relu(bn2(t2)); m_aggr += m (scatter); t3 = m @ pos_w1  (+stats)
    k_zero<<<1, 128, 0, stream>>>(ssum, 128);
    k_gemm_act_scatter<<<EB, 128, 0, stream>>>(bufB, scale, shift, wp1[l], bufA,
                                               ssum, ssqs, dst, maggr);
    k_bnfin<<<1, 64, 0, stream>>>(ssum, ssqs, pos_g + l * 64, pos_be + l * 64,
                                  (float)EE, scale, shift);
    // S7: s = relu(bn3(t3)) @ pos_w2 + b2; pacc[dst] += (p_i - p_j)*s
    k_pos_gate<<<EB, 128, 0, stream>>>(bufA, scale, shift, pos_w2 + l * 64,
                                       pos_b2 + l, p, src, dst, pacc);
    // S8: t4 = [h | m_aggr] @ upd_w1  (+stats)
    k_zero<<<1, 128, 0, stream>>>(ssum, 128);
    k_gemm_upd1<<<NB, 128, 0, stream>>>(h, maggr, wu1[l], bufA, ssum, ssqs);
    k_bnfin<<<1, 64, 0, stream>>>(ssum, ssqs, upd_g1 + l * 64, upd_be1 + l * 64,
                                  (float)NN, scale, shift);
    // S10: t5 = relu(bn4(t4)) @ upd_w2  (+stats)
    k_zero<<<1, 128, 0, stream>>>(ssum, 128);
    k_gemm_act<<<NB, 128, 0, stream>>>(bufA, scale, shift, wu2[l], bufB, ssum, ssqs, NN);
    k_bnfin<<<1, 64, 0, stream>>>(ssum, ssqs, upd_g2 + l * 64, upd_be2 + l * 64,
                                  (float)NN, scale, shift);
    // S12: h += relu(bn5(t5)); p += pacc/deg
    k_node_final<<<NB, 128, 0, stream>>>(bufB, scale, shift, h, p, pacc, deg);
  }

  // ---- pool + predict ----
  k_zero<<<gz(GG * 64 + GG), 256, 0, stream>>>(hg, GG * 64 + GG);
  k_pool<<<NB, 128, 0, stream>>>(h, batch, hg, cnt);
  k_out<<<1, 64, 0, stream>>>(hg, cnt, pred_w, pred_b, (float*)d_out);
}
